// MaskedAttention_2267742732943
// MI455X (gfx1250) — compile-verified, matrix operand order checked by reading, not run on hardware
//
#include <hip/hip_runtime.h>
#include <hip/hip_bf16.h>

typedef _Float16 half_t;
typedef __attribute__((ext_vector_type(16))) _Float16 v16h;
typedef __attribute__((ext_vector_type(8)))  _Float16 v8h;
typedef __attribute__((ext_vector_type(8)))  float    v8f;

#define BB 4
#define NQh 1024
#define HWk 4096
#define CC 1024
#define HH 8
#define DHd 128
#define MTOT (BB * HWk)   // 16384

__device__ __forceinline__ v8f zero_v8f() {
  v8f z;
#pragma unroll
  for (int k = 0; k < 8; ++k) z[k] = 0.0f;
  return z;
}

// Load a 16x32 f16 WMMA fragment (A or B operand) from a K-contiguous,
// row-major source. Caller passes p = base + row*ld + kbase + lanegroup*8.
// Lane pattern (ISA 7.12.2): lanes 0-15 hold K 0..7 (VGPR0-3) and K 16..23
// (VGPR4-7); lanes 16-31 hold K 8..15 and 24..31. Chunk2 is +16 halves.
__device__ __forceinline__ v16h load_frag(const half_t* p) {
  v8h lo = *(const v8h*)(p);
  v8h hi = *(const v8h*)(p + 16);
  v16h r;
#pragma unroll
  for (int k = 0; k < 8; ++k) { r[k] = lo[k]; r[k + 8] = hi[k]; }
  return r;
}

__device__ __forceinline__ v8f wmma_f16(v16h a, v16h b, v8f c) {
  return __builtin_amdgcn_wmma_f32_16x16x32_f16(false, a, false, b,
                                                (short)0, c, false, false);
}

__global__ void cvt_f32_to_f16(const float* __restrict__ in,
                               half_t* __restrict__ out, int n) {
  int idx = blockIdx.x * blockDim.x + threadIdx.x;
  if (idx < n) {
    // fp32 source is read exactly once: non-temporal load keeps L2 for the
    // f16 WMMA working set.
    out[idx] = (half_t)__builtin_nontemporal_load(in + idx);
  }
}

// out = A @ B^T + bias.  A: [M,K] f16 row-major, Bw: [N,K] f16 row-major.
// mode 0: f16 out[m*N+n]; mode 1: f32 out[m*N+n] (NT store); mode 2: f16
// out[n*M+m] (transposed).
// Block = 256 threads = 8 waves as 2(M) x 4(N); each wave does 64x64
// (16 WMMAs per 16 b128 fragment loads per K-step). Block tile = 128 x 256.
// M % 128 == 0, N % 256 == 0, K % 32 == 0 assumed.
__global__ void __launch_bounds__(256)
gemm_nt(const half_t* __restrict__ A, const half_t* __restrict__ Bw,
        const float* __restrict__ bias, void* __restrict__ outp,
        int M, int N, int K, int mode) {
  const int w = threadIdx.x >> 5;
  const int l = threadIdx.x & 31;
  const int g = l >> 4;       // lane group (K-chunk select)
  const int i = l & 15;       // row (A) / col (B) within 16-tile
  const int m0 = blockIdx.x * 128 + (w & 1) * 64;
  const int n0 = blockIdx.y * 256 + (w >> 1) * 64;

  v8f c[4][4];
#pragma unroll
  for (int mm = 0; mm < 4; ++mm)
#pragma unroll
    for (int nn = 0; nn < 4; ++nn) c[mm][nn] = zero_v8f();

  for (int k = 0; k < K; k += 32) {
    v16h af[4], bf[4];
#pragma unroll
    for (int mm = 0; mm < 4; ++mm)
      af[mm] = load_frag(A + (size_t)(m0 + mm * 16 + i) * K + k + g * 8);
#pragma unroll
    for (int nn = 0; nn < 4; ++nn)
      bf[nn] = load_frag(Bw + (size_t)(n0 + nn * 16 + i) * K + k + g * 8);
#pragma unroll
    for (int mm = 0; mm < 4; ++mm)
#pragma unroll
      for (int nn = 0; nn < 4; ++nn)
        c[mm][nn] = wmma_f16(af[mm], bf[nn], c[mm][nn]);
  }

  // Epilogue. C/D layout: VGPR r, lanes 0-15 -> M = r, lanes 16-31 -> M = r+8,
  // N = lane & 15.  Bias is always non-null: load unconditionally (no branch).
#pragma unroll
  for (int nn = 0; nn < 4; ++nn) {
    const int ng = n0 + nn * 16 + i;
    const float bv = bias[ng];
#pragma unroll
    for (int mm = 0; mm < 4; ++mm) {
#pragma unroll
      for (int r = 0; r < 8; ++r) {
        const int mg = m0 + mm * 16 + r + 8 * g;
        const float val = c[mm][nn][r] + bv;
        if (mode == 0) {
          ((half_t*)outp)[(size_t)mg * N + ng] = (half_t)val;
        } else if (mode == 1) {
          // final fp32 output: write-once, never re-read on device
          __builtin_nontemporal_store(val, &((float*)outp)[(size_t)mg * N + ng]);
        } else {
          ((half_t*)outp)[(size_t)ng * M + mg] = (half_t)val;
        }
      }
    }
  }
}

// Flash attention. Grid: B*H*(NQ/128) blocks, 256 threads = 8 waves,
// each wave owns a 16-query tile with full DH=128 accumulated in registers.
// Qf: [B,NQ,C] f16 ; Kf: [B,HW,C] f16 ; Vt: [C, B*HW] f16 (transposed) ;
// mask: [B,NQ,HW] f32 ; AO: [B,NQ,C] f16.
__global__ void __launch_bounds__(256)
flash_attn(const half_t* __restrict__ Qf, const half_t* __restrict__ Kf,
           const half_t* __restrict__ Vt, const float* __restrict__ mask,
           half_t* __restrict__ AO) {
  const float scale = 0.08838834764831845f;  // 1/sqrt(128)
  const int w = threadIdx.x >> 5;
  const int l = threadIdx.x & 31;
  const int g = l >> 4;
  const int i = l & 15;
  const int qblk = blockIdx.x & 7;
  const int bh = blockIdx.x >> 3;
  const int b = bh >> 3;     // H == 8
  const int h = bh & 7;
  const int qbase = qblk * 128 + w * 16;

  // Q fragments for the whole kernel: 16 queries x 128 d = 4 K-steps.
  v16h aQ[4];
  {
    const half_t* qrow =
        Qf + (size_t)(b * NQh + qbase + i) * CC + h * DHd + g * 8;
#pragma unroll
    for (int d = 0; d < 4; ++d) aQ[d] = load_frag(qrow + d * 32);
  }

  v8f O[8];
#pragma unroll
  for (int t = 0; t < 8; ++t) O[t] = zero_v8f();
  float mrow[8], lrow[8];
#pragma unroll
  for (int r = 0; r < 8; ++r) { mrow[r] = -1.0e30f; lrow[r] = 0.0f; }

  __shared__ __align__(16) half_t Pl[8 * 512];  // 1KB per wave: 16q x 32k f16
  half_t* myP = &Pl[w * 512];

  for (int kt = 0; kt < HWk; kt += 32) {
    // ---- scores: two 16x16 tiles over 32 keys, K-dim = d (4 steps of 32)
    v8f s0 = zero_v8f(), s1 = zero_v8f();
    const half_t* k0p =
        Kf + (size_t)(b * HWk + kt + i) * CC + h * DHd + g * 8;
    const half_t* k1p = k0p + (size_t)16 * CC;
#pragma unroll
    for (int d = 0; d < 4; ++d) {
      v16h bk0 = load_frag(k0p + d * 32);
      s0 = wmma_f16(aQ[d], bk0, s0);
      v16h bk1 = load_frag(k1p + d * 32);
      s1 = wmma_f16(aQ[d], bk1, s1);
    }

    // ---- scale + mask (C/D layout: row = r + 8*g, col = i); mask is
    // read-once fp32 -> non-temporal loads.
    float rm[8];
#pragma unroll
    for (int r = 0; r < 8; ++r) {
      const int qg = qbase + r + 8 * g;
      const float* mp = mask + ((size_t)b * NQh + qg) * HWk + kt + i;
      const float mk0 = __builtin_nontemporal_load(mp);
      const float mk1 = __builtin_nontemporal_load(mp + 16);
      const float v0 = (mk0 >= 0.5f) ? s0[r] * scale : -1.0e30f;
      const float v1 = (mk1 >= 0.5f) ? s1[r] * scale : -1.0e30f;
      s0[r] = v0; s1[r] = v1;
      rm[r] = fmaxf(v0, v1);
    }
    // row-max across the 16 lanes of each half-wave group
#pragma unroll
    for (int off = 1; off < 16; off <<= 1)
#pragma unroll
      for (int r = 0; r < 8; ++r)
        rm[r] = fmaxf(rm[r], __shfl_xor(rm[r], off, 32));

    // ---- online softmax update
    float alpha[8], ps[8];
#pragma unroll
    for (int r = 0; r < 8; ++r) {
      const float mn = fmaxf(mrow[r], rm[r]);
      alpha[r] = __expf(mrow[r] - mn);
      mrow[r] = mn;
      const float p0 = __expf(s0[r] - mn);
      const float p1 = __expf(s1[r] - mn);
      s0[r] = p0; s1[r] = p1;
      ps[r] = p0 + p1;
    }
#pragma unroll
    for (int off = 1; off < 16; off <<= 1)
#pragma unroll
      for (int r = 0; r < 8; ++r) ps[r] += __shfl_xor(ps[r], off, 32);
#pragma unroll
    for (int r = 0; r < 8; ++r) lrow[r] = lrow[r] * alpha[r] + ps[r];
#pragma unroll
    for (int t = 0; t < 8; ++t)
#pragma unroll
      for (int r = 0; r < 8; ++r) O[t][r] *= alpha[r];

    // ---- P: C/D layout -> LDS (absolute [q][key]) -> A-fragment layout
#pragma unroll
    for (int r = 0; r < 8; ++r) {
      myP[(r + 8 * g) * 32 + i]      = (half_t)s0[r];
      myP[(r + 8 * g) * 32 + 16 + i] = (half_t)s1[r];
    }
    asm volatile("s_wait_dscnt 0" ::: "memory");  // intra-wave RAW on LDS
    v16h aP = load_frag(myP + i * 32 + g * 8);

    // ---- PV: B operand from transposed V (key-contiguous rows)
    const half_t* vbase =
        Vt + (size_t)(h * DHd + i) * MTOT + b * HWk + kt + g * 8;
#pragma unroll
    for (int t = 0; t < 8; ++t) {
      v16h bv = load_frag(vbase + (size_t)(t * 16) * MTOT);
      O[t] = wmma_f16(aP, bv, O[t]);
    }
  }

  // ---- epilogue: normalize and store f16 [B,NQ,C]
#pragma unroll
  for (int r = 0; r < 8; ++r) {
    const float inv = 1.0f / lrow[r];
    const int qg = qbase + r + 8 * g;
#pragma unroll
    for (int t = 0; t < 8; ++t)
      AO[(size_t)(b * NQh + qg) * CC + h * DHd + t * 16 + i] =
          (half_t)(O[t][r] * inv);
  }
}

extern "C" void kernel_launch(void* const* d_in, const int* in_sizes, int n_in,
                              void* d_out, int out_size, void* d_ws, size_t ws_size,
                              hipStream_t stream) {
  const float* query  = (const float*)d_in[0];
  const float* memory = (const float*)d_in[1];
  const float* mask   = (const float*)d_in[2];
  const float* Wq = (const float*)d_in[3];
  const float* bq = (const float*)d_in[4];
  const float* Wk = (const float*)d_in[5];
  const float* bk = (const float*)d_in[6];
  const float* Wv = (const float*)d_in[7];
  const float* bv = (const float*)d_in[8];
  const float* Wo = (const float*)d_in[9];
  const float* bo = (const float*)d_in[10];
  float* out = (float*)d_out;

  // f16 workspace layout (halves). Total = 60M halves = 120 MB.
  const size_t nQin = (size_t)BB * NQh * CC;   // 4M
  const size_t nMem = (size_t)BB * HWk * CC;   // 16M
  const size_t nW   = (size_t)CC * CC;         // 1M
  half_t* ws   = (half_t*)d_ws;
  half_t* qf   = ws;            // query f16            (4M)
  half_t* memf = qf + nQin;     // memory f16           (16M)
  half_t* wqf  = memf + nMem;   // weights f16          (4x1M)
  half_t* wkf  = wqf + nW;
  half_t* wvf  = wkf + nW;
  half_t* wof  = wvf + nW;
  half_t* Qp   = wof + nW;      // Q projection         (4M)
  half_t* Kp   = Qp + nQin;     // K projection         (16M)
  half_t* Vtp  = Kp + nMem;     // V projection, transposed [C, B*HW] (16M)
  half_t* AO   = qf;            // attention out reuses query-f16 (dead by then)

  // 1) fp32 -> f16 conversions (NT loads of read-once fp32)
  cvt_f32_to_f16<<<(unsigned)(nQin / 256), 256, 0, stream>>>(query, qf, (int)nQin);
  cvt_f32_to_f16<<<(unsigned)(nMem / 256), 256, 0, stream>>>(memory, memf, (int)nMem);
  cvt_f32_to_f16<<<(unsigned)(nW / 256), 256, 0, stream>>>(Wq, wqf, (int)nW);
  cvt_f32_to_f16<<<(unsigned)(nW / 256), 256, 0, stream>>>(Wk, wkf, (int)nW);
  cvt_f32_to_f16<<<(unsigned)(nW / 256), 256, 0, stream>>>(Wv, wvf, (int)nW);
  cvt_f32_to_f16<<<(unsigned)(nW / 256), 256, 0, stream>>>(Wo, wof, (int)nW);

  // 2) projections (M = B*rows since Linear is batch-independent).
  //    Block tile 128(M) x 256(N).
  gemm_nt<<<dim3(4096 / 128, 1024 / 256), 256, 0, stream>>>(
      qf, wqf, bq, Qp, 4096, 1024, 1024, /*mode=*/0);
  gemm_nt<<<dim3(16384 / 128, 1024 / 256), 256, 0, stream>>>(
      memf, wkf, bk, Kp, 16384, 1024, 1024, /*mode=*/0);
  gemm_nt<<<dim3(16384 / 128, 1024 / 256), 256, 0, stream>>>(
      memf, wvf, bv, Vtp, 16384, 1024, 1024, /*mode=*/2);  // transposed store

  // 3) masked flash attention (B*H*(NQ/128) = 256 blocks)
  flash_attn<<<BB * HH * (NQh / 128), 256, 0, stream>>>(Qp, Kp, Vtp, mask, AO);

  // 4) output projection, fp32 out + bias (NT stores)
  gemm_nt<<<dim3(4096 / 128, 1024 / 256), 256, 0, stream>>>(
      AO, wof, bo, out, 4096, 1024, 1024, /*mode=*/1);
}
